// MultiHeadedAttention_12850542150154
// MI455X (gfx1250) — matrix operand order repaired and checked by reading, not compiled, not run on hardware
//
#include <hip/hip_runtime.h>
#include <hip/hip_bf16.h>
#include <math.h>

typedef __attribute__((ext_vector_type(16))) _Float16 v16h;
typedef __attribute__((ext_vector_type(8)))  _Float16 v8h;
typedef __attribute__((ext_vector_type(8)))  float    v8f;
typedef __attribute__((ext_vector_type(4)))  float    v4f;

#define TT 2048
#define SS 2048
#define BB 2
#define EE 1024
#define HH 16
#define ZZ 32   // B*H
#define DD 64

// Load 16 f32 (two 8-wide runs for A, one 16-wide run for B split in 4x float4),
// convert to f16 packed for the WMMA operand vector.
__device__ __forceinline__ v16h pack16_f32(const float* p0, const float* p1,
                                           const float* p2, const float* p3) {
  v4f a0 = *(const v4f*)p0, a1 = *(const v4f*)p1;
  v4f a2 = *(const v4f*)p2, a3 = *(const v4f*)p3;
  v16h r;
#pragma unroll
  for (int i = 0; i < 4; ++i) {
    r[i]      = (_Float16)a0[i];
    r[i + 4]  = (_Float16)a1[i];
    r[i + 8]  = (_Float16)a2[i];
    r[i + 12] = (_Float16)a3[i];
  }
  return r;
}

__device__ __forceinline__ v16h cat8h(const _Float16* lo, const _Float16* hi) {
  v8h l = *(const v8h*)lo;
  v8h h = *(const v8h*)hi;
  v16h r;
#pragma unroll
  for (int i = 0; i < 8; ++i) { r[i] = l[i]; r[i + 8] = h[i]; }
  return r;
}

// Y = (X @ W^T + bias) * scale, stored as f16 into per-head layouts.
// MODE 0/1: out[z][L][64]  (q with scale, k)   MODE 2: out[z][64][S] (v transposed)
// X: [M=4096][1024] f32 (rows = t*B+b), W: [1024][1024] f32 (row n = out feature)
template<int MODE>
__global__ __launch_bounds__(128) void qkv_proj_kernel(
    const float* __restrict__ X, const float* __restrict__ W,
    const float* __restrict__ bias, float scale, _Float16* __restrict__ out)
{
  const int lane = threadIdx.x & 31;
  const int wave = threadIdx.x >> 5;
  const int wid  = blockIdx.x * 4 + wave;      // 16384 tiles
  const int m_tile = wid >> 6;                  // 0..255
  const int n_tile = wid & 63;                  // 0..63
  const int hf  = lane >> 4;
  const int l16 = lane & 15;

  // A operand: row = lane%16, K = hf*8 + {0..7} and 16 + hf*8 + {0..7}
  const float* aptr = X + (size_t)(m_tile * 16 + l16) * EE + hf * 8;
  // B operand: col n = lane%16, K = hf*8 + {0..7} and 16 + hf*8 + {0..7} (B[k][n] = W[n][k])
  const int ncol = n_tile * 16 + l16;
  const float* bptr = W + (size_t)ncol * EE + hf * 8;

  v8f c = {};
#pragma unroll 4
  for (int kk = 0; kk < EE; kk += 32) {
    v16h a = pack16_f32(aptr + kk, aptr + kk + 4, aptr + kk + 16, aptr + kk + 20);
    v16h b = pack16_f32(bptr + kk, bptr + kk + 4, bptr + kk + 16, bptr + kk + 20);
    c = __builtin_amdgcn_wmma_f32_16x16x32_f16(false, a, false, b, (short)0, c, false, false);
  }
  const float bval = bias[ncol];
  const int h = ncol >> 6, d = ncol & 63;
#pragma unroll
  for (int i = 0; i < 8; ++i) {
    const int M = m_tile * 16 + hf * 8 + i;   // C layout: M = hf*8 + vgpr idx
    const int t = M >> 1, bidx = M & 1;       // B == 2
    const int z = bidx * HH + h;
    const float val = (c[i] + bval) * scale;
    if (MODE == 2)
      out[((size_t)z * DD + d) * SS + t] = (_Float16)val;
    else
      out[((size_t)z * TT + t) * DD + d] = (_Float16)val;
  }
}

__device__ __forceinline__ void store_ctx_tile(_Float16* __restrict__ ctx, v8f acc, v8f invv,
                                               int t0, int hf, int l16, int bidx, int h, int nt) {
  const int d = nt * 16 + l16;
#pragma unroll
  for (int i = 0; i < 8; ++i) {
    const int trow = t0 + hf * 8 + i;
    ctx[((size_t)trow * BB + bidx) * EE + h * DD + d] = (_Float16)(acc[i] * invv[i]);
  }
}

// Per (head z, 16-row t-tile): scores -> rowmax -> thresholded exp -> P@V.
__global__ __launch_bounds__(128) void attn_kernel(
    const _Float16* __restrict__ q_ws,  // [32][T][64]
    const _Float16* __restrict__ k_ws,  // [32][S][64]
    const _Float16* __restrict__ v_ws,  // [32][64][S]
    const int* __restrict__ mask,       // [T][S]
    float* __restrict__ wts,            // [32][T][S]
    float* __restrict__ inv_ws,         // [32][T]
    _Float16* __restrict__ ctx)         // [T][B][1024]
{
  __shared__ _Float16 lds_p[4][16 * 32];
  const int lane = threadIdx.x & 31;
  const int wave = threadIdx.x >> 5;
  const int job  = blockIdx.x * 4 + wave;     // 4096 jobs
  const int z  = job >> 7;
  const int t0 = (job & 127) * 16;
  const int hf = lane >> 4, l16 = lane & 15;

  // Preload Q tile as A operands for K=0..63 (two K-steps of 32)
  const _Float16* qp = q_ws + ((size_t)z * TT + t0 + l16) * DD + hf * 8;
  const v16h a_lo = cat8h(qp,      qp + 16);
  const v16h a_hi = cat8h(qp + 32, qp + 48);

  float* wrow = wts + (size_t)z * TT * SS;

  float lmax[8];
#pragma unroll
  for (int i = 0; i < 8; ++i) lmax[i] = -INFINITY;

  // ---- Pass A: raw scores + running row max ----
  for (int st = 0; st < SS / 16; ++st) {
    const int s0 = st * 16;
    // B operand: col n = s0+l16 (a K row of scores), K = head dim: hf*8+{0..7}, 16+hf*8+{0..7}
    const _Float16* kp = k_ws + ((size_t)z * SS + s0 + l16) * DD + hf * 8;
    const v16h b_lo = cat8h(kp,      kp + 16);
    const v16h b_hi = cat8h(kp + 32, kp + 48);
    v8f c = {};
    c = __builtin_amdgcn_wmma_f32_16x16x32_f16(false, a_lo, false, b_lo, (short)0, c, false, false);
    c = __builtin_amdgcn_wmma_f32_16x16x32_f16(false, a_hi, false, b_hi, (short)0, c, false, false);
#pragma unroll
    for (int i = 0; i < 8; ++i) {
      const int trow = t0 + hf * 8 + i;
      const int scol = s0 + l16;
      float v = c[i];
      if (mask[(size_t)trow * SS + scol] == 0) v = -INFINITY;
      wrow[(size_t)trow * SS + scol] = v;
      lmax[i] = fmaxf(lmax[i], v);
    }
  }
  // reduce row max across the 16 lanes of each half-wave (xor masks stay in-half)
  float rowmax[8];
#pragma unroll
  for (int i = 0; i < 8; ++i) {
    float m = lmax[i];
    m = fmaxf(m, __shfl_xor(m, 1, 32));
    m = fmaxf(m, __shfl_xor(m, 2, 32));
    m = fmaxf(m, __shfl_xor(m, 4, 32));
    m = fmaxf(m, __shfl_xor(m, 8, 32));
    rowmax[i] = m;
  }

  // ---- Pass B: thresholded exp (unnormalized) + P@V ----
  const float DTH = 3.8918202981106265f;   // log(2000/40 - 1)
  float lsum[8] = {0, 0, 0, 0, 0, 0, 0, 0};
  v8f acc0 = {}, acc1 = {}, acc2 = {}, acc3 = {};
  _Float16* slab = lds_p[wave];

  for (int sp = 0; sp < SS / 32; ++sp) {
    const int s0 = sp * 32;
#pragma unroll
    for (int sub = 0; sub < 2; ++sub) {
      const int ss = s0 + sub * 16;
#pragma unroll
      for (int i = 0; i < 8; ++i) {
        const int trow = t0 + hf * 8 + i;
        const int scol = ss + l16;
        const float sc = wrow[(size_t)trow * SS + scol];
        const float dd = sc - rowmax[i];
        const float e = (dd >= -DTH) ? __expf(dd) : 0.0f;
        lsum[i] += e;
        wrow[(size_t)trow * SS + scol] = e;
        slab[(hf * 8 + i) * 32 + sub * 16 + l16] = (_Float16)e;
      }
    }
    __syncthreads();   // uniform trip count across all 4 waves
    // Re-read P from LDS in A-operand layout: row=l16, K = hf*16+{0..15}
    const _Float16* pp = slab + l16 * 32 + hf * 16;
    const v16h pa = cat8h(pp, pp + 8);
    // V B operand: col d = nt*16+l16, K = s index (contiguous: v stored [z][d][S])
    const _Float16* vb0 = v_ws + ((size_t)z * DD + l16) * SS + s0 + hf * 16;
    const _Float16* vb1 = vb0 + (size_t)16 * SS;
    const _Float16* vb2 = vb0 + (size_t)32 * SS;
    const _Float16* vb3 = vb0 + (size_t)48 * SS;
    acc0 = __builtin_amdgcn_wmma_f32_16x16x32_f16(false, pa, false, cat8h(vb0, vb0 + 8), (short)0, acc0, false, false);
    acc1 = __builtin_amdgcn_wmma_f32_16x16x32_f16(false, pa, false, cat8h(vb1, vb1 + 8), (short)0, acc1, false, false);
    acc2 = __builtin_amdgcn_wmma_f32_16x16x32_f16(false, pa, false, cat8h(vb2, vb2 + 8), (short)0, acc2, false, false);
    acc3 = __builtin_amdgcn_wmma_f32_16x16x32_f16(false, pa, false, cat8h(vb3, vb3 + 8), (short)0, acc3, false, false);
    __syncthreads();
  }

  // ---- Epilogue: rowsum -> 1/sum, write inv and f16 context ----
  v8f invv;
#pragma unroll
  for (int i = 0; i < 8; ++i) {
    float s = lsum[i];
    s += __shfl_xor(s, 1, 32);
    s += __shfl_xor(s, 2, 32);
    s += __shfl_xor(s, 4, 32);
    s += __shfl_xor(s, 8, 32);
    invv[i] = 1.0f / s;    // >= 1 element kept (the max itself), so s >= 1
  }
  if (l16 == 0) {
#pragma unroll
    for (int i = 0; i < 8; ++i)
      inv_ws[(size_t)z * TT + t0 + hf * 8 + i] = invv[i];
  }
  const int bidx = z >> 4, h = z & 15;
  store_ctx_tile(ctx, acc0, invv, t0, hf, l16, bidx, h, 0);
  store_ctx_tile(ctx, acc1, invv, t0, hf, l16, bidx, h, 1);
  store_ctx_tile(ctx, acc2, invv, t0, hf, l16, bidx, h, 2);
  store_ctx_tile(ctx, acc3, invv, t0, hf, l16, bidx, h, 3);
}

// weights[z][t][s] *= inv_ws[z*T+t]   (float4 vectorized; S multiple of 4)
__global__ void norm_kernel(float* __restrict__ wts, const float* __restrict__ inv_ws) {
  const size_t total4 = (size_t)ZZ * TT * SS / 4;
  const size_t stride = (size_t)gridDim.x * blockDim.x;
  for (size_t idx = (size_t)blockIdx.x * blockDim.x + threadIdx.x; idx < total4; idx += stride) {
    v4f v = ((v4f*)wts)[idx];
    const float inv = inv_ws[(idx * 4) >> 11];   // S == 2048
    v *= inv;
    ((v4f*)wts)[idx] = v;
  }
}

// out = ctx(f16) @ Wo^T + bo, f32 output, rows already in [t][b] order.
__global__ __launch_bounds__(128) void out_proj_kernel(
    const _Float16* __restrict__ ctx, const float* __restrict__ W,
    const float* __restrict__ bias, float* __restrict__ out)
{
  const int lane = threadIdx.x & 31;
  const int wave = threadIdx.x >> 5;
  const int wid  = blockIdx.x * 4 + wave;
  const int m_tile = wid >> 6;
  const int n_tile = wid & 63;
  const int hf = lane >> 4, l16 = lane & 15;

  const _Float16* aptr = ctx + (size_t)(m_tile * 16 + l16) * EE + hf * 16;
  const int ncol = n_tile * 16 + l16;
  const float* bptr = W + (size_t)ncol * EE + hf * 16;

  v8f c = {};
#pragma unroll 4
  for (int kk = 0; kk < EE; kk += 32) {
    v16h a = cat8h(aptr + kk, aptr + kk + 8);
    v16h b = pack16_f32(bptr + kk, bptr + kk + 4, bptr + kk + 8, bptr + kk + 12);
    c = __builtin_amdgcn_wmma_f32_16x16x32_f16(false, a, false, b, (short)0, c, false, false);
  }
  const float bval = bias[ncol];
#pragma unroll
  for (int i = 0; i < 8; ++i) {
    const int M = m_tile * 16 + hf * 8 + i;
    out[(size_t)M * EE + ncol] = c[i] + bval;
  }
}

extern "C" void kernel_launch(void* const* d_in, const int* in_sizes, int n_in,
                              void* d_out, int out_size, void* d_ws, size_t ws_size,
                              hipStream_t stream) {
  (void)in_sizes; (void)n_in; (void)out_size; (void)ws_size;
  const float* query = (const float*)d_in[0];
  const float* key   = (const float*)d_in[1];
  const float* value = (const float*)d_in[2];
  const int*   mask  = (const int*)d_in[3];
  const float* Wq = (const float*)d_in[4];
  const float* bq = (const float*)d_in[5];
  const float* Wk = (const float*)d_in[6];
  const float* bk = (const float*)d_in[7];
  const float* Wv = (const float*)d_in[8];
  const float* bv = (const float*)d_in[9];
  const float* Wo = (const float*)d_in[10];
  const float* bo = (const float*)d_in[11];

  char* ws = (char*)d_ws;
  _Float16* q_ws = (_Float16*)(ws);                          // 8 MB  [32][T][64]
  _Float16* k_ws = (_Float16*)(ws + ((size_t)8  << 20));     // 8 MB  [32][S][64]
  _Float16* v_ws = (_Float16*)(ws + ((size_t)16 << 20));     // 8 MB  [32][64][S]
  _Float16* ctx  = (_Float16*)(ws + ((size_t)24 << 20));     // 8 MB  [T][B][1024]
  float*    inv_ws = (float*)(ws + ((size_t)32 << 20));      // 256 KB [32][T]

  float* attn_out = (float*)d_out;                           // [T][B][E]
  float* wts = attn_out + (size_t)TT * BB * EE;              // [32][T][S]

  const float scaling = 0.125f;   // DK^-0.5, DK = 64
  dim3 blk(128);
  qkv_proj_kernel<0><<<4096, blk, 0, stream>>>(query, Wq, bq, scaling, q_ws);
  qkv_proj_kernel<1><<<4096, blk, 0, stream>>>(key,   Wk, bk, 1.0f,    k_ws);
  qkv_proj_kernel<2><<<4096, blk, 0, stream>>>(value, Wv, bv, 1.0f,    v_ws);
  attn_kernel<<<1024, blk, 0, stream>>>(q_ws, k_ws, v_ws, mask, wts, inv_ws, ctx);
  norm_kernel<<<8192, 256, 0, stream>>>(wts, inv_ws);
  out_proj_kernel<<<4096, blk, 0, stream>>>(ctx, Wo, bo, attn_out);
}
